// NUFFTCascade_38706245272064
// MI455X (gfx1250) — matrix operand order repaired and checked by reading, not compiled, not run on hardware
//
#include <hip/hip_runtime.h>
#include <hip/hip_bf16.h>
#include <math.h>

#define H_ 1024
#define W_ 1024
#define HW_ (H_ * W_)
#define M_ 524288
#define PITER_ 10
#define NPCG_ 4

typedef __attribute__((ext_vector_type(16))) _Float16 v16h;
typedef __attribute__((ext_vector_type(8)))  float    v8f;

// ---------------- utility kernels ----------------
__global__ void k_zero_f(float* p, int n) {
  int i = blockIdx.x * blockDim.x + threadIdx.x;
  if (i < n) p[i] = 0.f;
}

__global__ void k_init_state(const float* __restrict__ xr, const float* __restrict__ xi,
                             float2* __restrict__ x0, float2* __restrict__ v) {
  int i = blockIdx.x * blockDim.x + threadIdx.x;
  if (i < HW_) {
    x0[i] = make_float2(xr[i], xi[i]);
    v[i]  = make_float2(1.f, 0.f);
  }
}

// bilinear interp indices/weights (AoS int4/float4 for single-b128 access in
// the bandwidth-bound gather/scatter loops) + radius max via bit-atomic-max
__global__ void k_interp(const float* __restrict__ ktraj, int4* __restrict__ lin,
                         float4* __restrict__ wts, float* rmax_bits) {
  int m = blockIdx.x * blockDim.x + threadIdx.x;
  if (m >= M_) return;
  float kx = ktraj[2 * m], ky = ktraj[2 * m + 1];
  float gx = kx * 1024.f, gy = ky * 1024.f;
  float fx = floorf(gx), fy = floorf(gy);
  float tx = gx - fx, ty = gy - fy;
  int ix = (int)fx, iy = (int)fy;
  int ix0 = ix & 1023, ix1 = (ix + 1) & 1023;
  int iy0 = iy & 1023, iy1 = (iy + 1) & 1023;
  lin[m] = make_int4(ix0 * W_ + iy0, ix1 * W_ + iy0, ix0 * W_ + iy1, ix1 * W_ + iy1);
  wts[m] = make_float4((1.f - tx) * (1.f - ty), tx * (1.f - ty), (1.f - tx) * ty, tx * ty);
  float dx = kx - 0.5f, dy = ky - 0.5f;
  float r = sqrtf(dx * dx + dy * dy);
  atomicMax((unsigned int*)rmax_bits, __float_as_uint(r));
}

__global__ void k_dcf(const float* __restrict__ ktraj, const float* rmax_bits,
                      float* __restrict__ sdcf) {
  int m = blockIdx.x * blockDim.x + threadIdx.x;
  if (m >= M_) return;
  float dx = ktraj[2 * m] - 0.5f, dy = ktraj[2 * m + 1] - 0.5f;
  float r = sqrtf(dx * dx + dy * dy);
  float rmax = __uint_as_float(*(const unsigned int*)rmax_bits);
  sdcf[m] = sqrtf(r / (rmax + 1e-8f));
}

// ---------------- 1024-point Stockham FFT (one line per workgroup) ----------------
// sgn = -1: forward (exp(-i..)), sgn = +1: inverse. Ortho scale 1/32 per 1D pass.
__global__ __launch_bounds__(256) void k_fft1d(const float2* __restrict__ in,
                                               float2* __restrict__ out,
                                               float sgn, int rowpass) {
  __shared__ float2 bufA[1024];
  __shared__ float2 bufB[1024];
  int line = blockIdx.x;
  int t = threadIdx.x;
  for (int i = t; i < 1024; i += 256) {
    int g = rowpass ? (line * 1024 + i) : (i * 1024 + line);
    bufA[i] = in[g];
  }
  __syncthreads();
  float2* src = bufA;
  float2* dst = bufB;
  for (int stage = 0; stage < 10; ++stage) {
    int mm = 1 << stage;
    int ll = 512 >> stage;
    for (int u = t; u < 512; u += 256) {
      int j = u >> stage;
      int k = u & (mm - 1);
      float sv, cv;
      float ang = sgn * 3.14159265358979f * (float)j / (float)ll;
      __sincosf(ang, &sv, &cv);
      float2 a = src[k + j * mm];
      float2 b = src[k + j * mm + 512];
      float2 sum = make_float2(a.x + b.x, a.y + b.y);
      float2 dif = make_float2(a.x - b.x, a.y - b.y);
      float2 wd  = make_float2(cv * dif.x - sv * dif.y, cv * dif.y + sv * dif.x);
      dst[k + 2 * j * mm]      = sum;
      dst[k + 2 * j * mm + mm] = wd;
    }
    __syncthreads();
    float2* tmp = src; src = dst; dst = tmp;
  }
  const float scale = 0.03125f; // 1/sqrt(1024)
  for (int i = t; i < 1024; i += 256) {
    int g = rowpass ? (line * 1024 + i) : (i * 1024 + line);
    out[g] = make_float2(src[i].x * scale, src[i].y * scale);
  }
}

// ---------------- NUFFT sampling (gather) with optional k-space residual ----------------
__global__ void k_gather(const float2* __restrict__ F, const int4* __restrict__ lin,
                         const float4* __restrict__ wts, float2* __restrict__ av,
                         const float* __restrict__ kr, const float* __restrict__ ki,
                         const float* __restrict__ sdcf, int sub) {
  int m = blockIdx.x * blockDim.x + threadIdx.x;
  if (m >= M_) return;
  int4   l = lin[m];
  float4 w = wts[m];
  float2 f0 = F[l.x], f1 = F[l.y], f2 = F[l.z], f3 = F[l.w];
  float2 acc;
  acc.x = f0.x * w.x + f1.x * w.y + f2.x * w.z + f3.x * w.w;
  acc.y = f0.y * w.x + f1.y * w.y + f2.y * w.z + f3.y * w.w;
  if (sub) {
    float s = sdcf[m];
    acc.x -= kr[m] * s;
    acc.y -= ki[m] * s;
  }
  av[m] = acc;
}

// ---------------- gridding (scatter-add) via hardware FP32 atomics ----------------
__global__ void k_scatter(const float2* __restrict__ kv, const int4* __restrict__ lin,
                          const float4* __restrict__ wts, float2* __restrict__ grid) {
  int m = blockIdx.x * blockDim.x + threadIdx.x;
  if (m >= M_) return;
  float2 d = kv[m];
  int4   l = lin[m];
  float4 w = wts[m];
  unsafeAtomicAdd(&grid[l.x].x, w.x * d.x);
  unsafeAtomicAdd(&grid[l.x].y, w.x * d.y);
  unsafeAtomicAdd(&grid[l.y].x, w.y * d.x);
  unsafeAtomicAdd(&grid[l.y].y, w.y * d.y);
  unsafeAtomicAdd(&grid[l.z].x, w.z * d.x);
  unsafeAtomicAdd(&grid[l.z].y, w.z * d.y);
  unsafeAtomicAdd(&grid[l.w].x, w.w * d.x);
  unsafeAtomicAdd(&grid[l.w].y, w.w * d.y);
}

// ---------------- squared-norm reduction ----------------
__global__ __launch_bounds__(256) void k_norm2(const float2* __restrict__ z, int n, float* acc) {
  __shared__ float red[256];
  int tid = threadIdx.x;
  float s = 0.f;
  for (int i = blockIdx.x * blockDim.x + tid; i < n; i += gridDim.x * blockDim.x) {
    float2 t = z[i];
    s += t.x * t.x + t.y * t.y;
  }
  red[tid] = s;
  __syncthreads();
  for (int o = 128; o > 0; o >>= 1) {
    if (tid < o) red[tid] += red[tid + o];
    __syncthreads();
  }
  if (tid == 0) unsafeAtomicAdd(acc, red[0]);
}

__global__ void k_scale_invnorm(float2* __restrict__ v, const float* s) {
  int i = blockIdx.x * blockDim.x + threadIdx.x;
  if (i >= HW_) return;
  float inv = 1.f / (sqrtf(*s) + 1e-12f);
  v[i].x *= inv;
  v[i].y *= inv;
}

__global__ void k_finalize(const float* s_av, const float* s_v, const float* w_raw, float* wreg) {
  float nrm = sqrtf(*s_av) / (sqrtf(*s_v) + 1e-12f);
  float sig = 1.f / (1.f + __expf(-w_raw[0]));
  *wreg = 2.f * sig / (nrm * nrm);
}

// ---------------- fused CNN (2->32->2, 3x3) as implicit GEMM on WMMA + PGD update ----------
// One workgroup (4 waves) computes a 16x16 output tile.
//  stage:  20x20 input tile -> LDS via gfx1250 async copy (ASYNCcnt path)
//  GEMM1:  hidden(18x18,32ch) = im2col(x, K=18 pad 32) x W1 -> LDS (f16)
//  GEMM2:  cnn(16x16,2ch)     = im2col(hidden, K=288=9*32) x W2
//  xout = xin - w_reg*res - cnn
__global__ __launch_bounds__(128) void k_conv_update(
    const float2* __restrict__ xin, const float2* __restrict__ res,
    const float* __restrict__ w1, const float* __restrict__ b1,
    const float* __restrict__ w2, const float* __restrict__ b2,
    const float* __restrict__ wreg_p, float2* __restrict__ xout) {
  __shared__ float2   xtile[20][21];    // 20x20 input tile (+1 col pad), zero border
  __shared__ _Float16 hid[336][36];     // 18x18 hidden tile +slack rows, padded channels

  int bx  = blockIdx.x;
  int tx0 = (bx & 63) * 16;
  int ty0 = (bx >> 6) * 16;
  int tid  = threadIdx.x;
  int wave = tid >> 5;
  int lane = tid & 31;
  int hiHalf = lane >> 4;   // 0: lanes 0-15, 1: lanes 16-31

  // ---- stage input tile into LDS with async global->LDS copies ----
  for (int i = tid; i < 400; i += 128) {
    int r = i / 20, c = i % 20;
    int iy = ty0 - 2 + r, ix = tx0 - 2 + c;
    int iyc = min(max(iy, 0), H_ - 1);          // clamp so OOB lanes still load legally
    int ixc = min(max(ix, 0), W_ - 1);
    const float2* gsrc = xin + ((size_t)iyc * W_ + ixc);
    unsigned ldso = (unsigned)(size_t)(void*)&xtile[r][c];
    asm volatile("global_load_async_to_lds_b64 %0, %1, off"
                 :: "v"(ldso), "v"(gsrc) : "memory");
  }
  asm volatile("s_wait_asynccnt 0" ::: "memory");
  // zero-fill the out-of-image border cells ('SAME' conv zero padding)
  for (int i = tid; i < 400; i += 128) {
    int r = i / 20, c = i % 20;
    int iy = ty0 - 2 + r, ix = tx0 - 2 + c;
    if (iy < 0 || iy >= H_ || ix < 0 || ix >= W_) xtile[r][c] = make_float2(0.f, 0.f);
  }
  __syncthreads();

  // ---- B fragments for GEMM1 (K=32 padded, N-tiles 0 and 1) ----
  // B layout (16-bit, 32x16): n = lane&15, element e -> K = 16*(e>=8) + 8*hiHalf + (e&7)
  v16h B0, B1;
  {
    int n = lane & 15;
#pragma unroll
    for (int e = 0; e < 16; ++e) {
      int k = ((e & 8) << 1) + (hiHalf << 3) + (e & 7);
      float wv0 = 0.f, wv1 = 0.f;
      if (k < 18) {
        int ic = k / 9, tap = k % 9;
        wv0 = w1[n * 18 + ic * 9 + tap];
        wv1 = w1[(n + 16) * 18 + ic * 9 + tap];
      }
      B0[e] = (_Float16)wv0;
      B1[e] = (_Float16)wv1;
    }
  }

  // ---- GEMM1: 21 M-tiles of 16 hidden pixels (324 valid), round-robin over waves ----
  for (int mt = wave; mt < 21; mt += 4) {
    int hp = min(mt * 16 + (lane & 15), 323);   // clamp: rows >=324 are write-only slack
    int hy = hp / 18, hx = hp % 18;
    // A layout (16-bit, 16x32): m = lane&15; elem e: v=e>>1,
    // K = 16*(v>=4) + 8*hiHalf + 2*(v&3) + (e&1)
    v16h A;
#pragma unroll
    for (int e = 0; e < 16; ++e) {
      int vv = e >> 1;
      int k = ((vv & 4) << 2) + (hiHalf << 3) + ((vv & 3) << 1) + (e & 1);
      int ic = k / 9, tap = k - ic * 9;         // tap in [0,9)
      float2 p = xtile[hy + tap / 3][hx + tap % 3];   // always in-range
      float val = (k < 18) ? ((ic == 0) ? p.x : p.y) : 0.f;
      A[e] = (_Float16)val;
    }
    v8f z = {0.f, 0.f, 0.f, 0.f, 0.f, 0.f, 0.f, 0.f};
    v8f c0 = __builtin_amdgcn_wmma_f32_16x16x32_f16(false, A, false, B0, (short)0, z, false, false);
    v8f c1 = __builtin_amdgcn_wmma_f32_16x16x32_f16(false, A, false, B1, (short)0, z, false, false);
    // C/D layout: n = lane&15, m = r + 8*hiHalf  (store unguarded into slack rows)
    int nc = lane & 15;
    float bn0 = b1[nc], bn1 = b1[nc + 16];
#pragma unroll
    for (int r = 0; r < 8; ++r) {
      int hp2 = mt * 16 + r + (hiHalf << 3);    // < 336 always
      hid[hp2][nc]      = (_Float16)(c0[r] + bn0);
      hid[hp2][nc + 16] = (_Float16)(c1[r] + bn1);
    }
  }
  __syncthreads();

  // ---- GEMM2: 16 M-tiles (one output row each), K = 9 taps x 32 channels ----
  v8f acc0 = {0.f, 0.f, 0.f, 0.f, 0.f, 0.f, 0.f, 0.f};
  v8f acc1 = acc0, acc2 = acc0, acc3 = acc0;
  for (int tap = 0; tap < 9; ++tap) {
    v16h B2;
    {
      int n = lane & 15;
#pragma unroll
      for (int e = 0; e < 16; ++e) {
        int c = ((e & 8) << 1) + (hiHalf << 3) + (e & 7);
        float wv = 0.f;
        if (n < 2) wv = w2[n * 288 + c * 9 + tap];
        B2[e] = (_Float16)wv;
      }
    }
    int dy = tap / 3, dx = tap % 3;
#pragma unroll
    for (int q = 0; q < 4; ++q) {
      int oy = wave * 4 + q;                // output row within tile == M-tile index
      int ox = lane & 15;
      int hrow = (oy + dy) * 18 + (ox + dx);
      v16h A2;
#pragma unroll
      for (int e = 0; e < 16; ++e) {
        int vv = e >> 1;
        int c = ((vv & 4) << 2) + (hiHalf << 3) + ((vv & 3) << 1) + (e & 1);
        A2[e] = hid[hrow][c];
      }
      if (q == 0) acc0 = __builtin_amdgcn_wmma_f32_16x16x32_f16(false, A2, false, B2, (short)0, acc0, false, false);
      if (q == 1) acc1 = __builtin_amdgcn_wmma_f32_16x16x32_f16(false, A2, false, B2, (short)0, acc1, false, false);
      if (q == 2) acc2 = __builtin_amdgcn_wmma_f32_16x16x32_f16(false, A2, false, B2, (short)0, acc2, false, false);
      if (q == 3) acc3 = __builtin_amdgcn_wmma_f32_16x16x32_f16(false, A2, false, B2, (short)0, acc3, false, false);
    }
  }

  // ---- writeback + PGD update: x' = x - w_reg*res - cnn ----
  float wreg = *wreg_p;
  int n = lane & 15;            // channel (0=real, 1=imag valid)
  if (n < 2) {
    float bn = b2[n];
#pragma unroll
    for (int q = 0; q < 4; ++q) {
      v8f acc = (q == 0) ? acc0 : (q == 1) ? acc1 : (q == 2) ? acc2 : acc3;
      int oy = wave * 4 + q;
#pragma unroll
      for (int r = 0; r < 8; ++r) {
        int ox = r + (hiHalf << 3);   // D: m = r + 8*hiHalf
        int gp = (ty0 + oy) * W_ + (tx0 + ox);
        float cnn = acc[r] + bn;
        float2 xv = xin[gp];
        float2 rv = res[gp];
        float xc = (n == 0) ? xv.x : xv.y;
        float rc = (n == 0) ? rv.x : rv.y;
        ((float*)&xout[gp])[n] = xc - wreg * rc - cnn;
      }
    }
  }
}

__global__ void k_output(const float2* __restrict__ x, float* __restrict__ out) {
  int i = blockIdx.x * blockDim.x + threadIdx.x;
  if (i < HW_) {
    out[i]       = x[i].x;
    out[HW_ + i] = x[i].y;
  }
}

// ---------------- host orchestration ----------------
extern "C" void kernel_launch(void* const* d_in, const int* in_sizes, int n_in,
                              void* d_out, int out_size, void* d_ws, size_t ws_size,
                              hipStream_t stream) {
  (void)in_sizes; (void)n_in; (void)out_size; (void)ws_size;
  const float* xr    = (const float*)d_in[0];
  const float* xi    = (const float*)d_in[1];
  const float* kr    = (const float*)d_in[2];
  const float* ki    = (const float*)d_in[3];
  const float* ktraj = (const float*)d_in[4];
  const float* w_raw = (const float*)d_in[5];
  const float* w1    = (const float*)d_in[6];
  const float* b1    = (const float*)d_in[7];
  const float* w2    = (const float*)d_in[8];
  const float* b2    = (const float*)d_in[9];

  char* ws = (char*)d_ws;
  size_t off = 0;
  auto alloc = [&](size_t bytes) -> void* {
    void* p = ws + off;
    off = (off + bytes + 255) & ~(size_t)255;
    return p;
  };
  float*  scal = (float*)alloc(256);                  // [0]=rmax bits [1]=s_av [2]=s_v [3]=s_n [4]=w_reg
  int4*   lin  = (int4*)alloc((size_t)M_ * 16);
  float4* wts  = (float4*)alloc((size_t)M_ * 16);
  float*  sdcf = (float*)alloc((size_t)M_ * 4);
  float2* G    = (float2*)alloc((size_t)HW_ * 8);     // FFT ping
  float2* F    = (float2*)alloc((size_t)HW_ * 8);     // FFT pong
  float2* v    = (float2*)alloc((size_t)HW_ * 8);
  float2* kv   = (float2*)alloc((size_t)M_ * 8);
  float2* res  = (float2*)alloc((size_t)HW_ * 8);
  float2* x0   = (float2*)alloc((size_t)HW_ * 8);
  float2* x1   = (float2*)alloc((size_t)HW_ * 8);

  const int BM = M_ / 256;    // 2048
  const int BP = HW_ / 256;   // 4096

  k_zero_f<<<1, 64, 0, stream>>>(scal, 8);
  k_interp<<<BM, 256, 0, stream>>>(ktraj, lin, wts, scal + 0);
  k_dcf<<<BM, 256, 0, stream>>>(ktraj, scal + 0, sdcf);
  k_init_state<<<BP, 256, 0, stream>>>(xr, xi, x0, v);

  // ---- power iteration for operator norm ----
  for (int it = 0; it < PITER_; ++it) {
    k_fft1d<<<1024, 256, 0, stream>>>(v, G, -1.f, 1);
    k_fft1d<<<1024, 256, 0, stream>>>(G, F, -1.f, 0);
    k_gather<<<BM, 256, 0, stream>>>(F, lin, wts, kv, nullptr, nullptr, nullptr, 0);
    k_zero_f<<<1, 64, 0, stream>>>(scal + 1, 3);
    k_norm2<<<512, 256, 0, stream>>>(kv, M_, scal + 1);
    k_norm2<<<512, 256, 0, stream>>>(v, HW_, scal + 2);
    k_zero_f<<<2 * BP, 256, 0, stream>>>((float*)G, 2 * HW_);
    k_scatter<<<BM, 256, 0, stream>>>(kv, lin, wts, G);
    k_fft1d<<<1024, 256, 0, stream>>>(G, F, +1.f, 1);
    k_fft1d<<<1024, 256, 0, stream>>>(F, v, +1.f, 0);
    k_norm2<<<512, 256, 0, stream>>>(v, HW_, scal + 3);
    k_scale_invnorm<<<BP, 256, 0, stream>>>(v, scal + 3);
  }
  k_finalize<<<1, 1, 0, stream>>>(scal + 1, scal + 2, w_raw, scal + 4);

  // ---- npcg PGD iterations ----
  float2* xa = x0;
  float2* xb = x1;
  for (int it = 0; it < NPCG_; ++it) {
    k_fft1d<<<1024, 256, 0, stream>>>(xa, G, -1.f, 1);
    k_fft1d<<<1024, 256, 0, stream>>>(G, F, -1.f, 0);
    k_gather<<<BM, 256, 0, stream>>>(F, lin, wts, kv, kr, ki, sdcf, 1);
    k_zero_f<<<2 * BP, 256, 0, stream>>>((float*)G, 2 * HW_);
    k_scatter<<<BM, 256, 0, stream>>>(kv, lin, wts, G);
    k_fft1d<<<1024, 256, 0, stream>>>(G, F, +1.f, 1);
    k_fft1d<<<1024, 256, 0, stream>>>(F, res, +1.f, 0);
    k_conv_update<<<4096, 128, 0, stream>>>(xa, res, w1, b1, w2, b2, scal + 4, xb);
    float2* t = xa; xa = xb; xb = t;
  }
  k_output<<<BP, 256, 0, stream>>>(xa, (float*)d_out);
}
